// EdgeUpdate2D_87522843558205
// MI455X (gfx1250) — matrix operand order repaired and checked by reading, not compiled, not run on hardware
//
#include <hip/hip_runtime.h>

typedef __attribute__((ext_vector_type(16))) _Float16 v16h;
typedef __attribute__((ext_vector_type(8)))  _Float16 v8h;
typedef __attribute__((ext_vector_type(8)))  float    v8f;
typedef __attribute__((ext_vector_type(4)))  float    v4f;

#define D_NODE 64
#define D_EDGE 64
#define IN_DIM 192
#define EDGES_PER_BLOCK 128
#define H1_LD 72  // padded f16 row stride for per-wave h1 tile (16B-aligned rows)

// Build a 16-f16 A-fragment from two 8-float chunks (K ascending per ISA layout).
__device__ __forceinline__ v16h pack2(const float* p0, const float* p1) {
  v4f a0 = *(const v4f*)p0;
  v4f a1 = *(const v4f*)(p0 + 4);
  v4f b0 = *(const v4f*)p1;
  v4f b1 = *(const v4f*)(p1 + 4);
  v16h r;
#pragma unroll
  for (int j = 0; j < 4; ++j) {
    r[j]      = (_Float16)a0[j];
    r[4 + j]  = (_Float16)a1[j];
    r[8 + j]  = (_Float16)b0[j];
    r[12 + j] = (_Float16)b1[j];
  }
  return r;
}

__device__ __forceinline__ float silu_f(float x) {
  return x / (1.0f + __expf(-x));
}

__global__ __launch_bounds__(256)
void edge_mlp_ln_kernel(const float* __restrict__ node_scalars,
                        const float* __restrict__ edge_feats,
                        const int*   __restrict__ src_idxs,
                        const int*   __restrict__ dst_idxs,
                        const float* __restrict__ W1,
                        const float* __restrict__ b1,
                        const float* __restrict__ W2,
                        const float* __restrict__ b2,
                        const float* __restrict__ ln_gamma,
                        const float* __restrict__ ln_beta,
                        float* __restrict__ out,
                        int E) {
  __shared__ _Float16 sW1[D_EDGE * IN_DIM];   // W1^T as [n][k], f16
  __shared__ _Float16 sW2[D_EDGE * D_EDGE];   // W2^T as [n][k], f16
  __shared__ _Float16 sH1[8 * 16 * H1_LD];    // per-wave 16x64 f16 h1 tile

  const int tid  = threadIdx.x;
  const int wave = tid >> 5;
  const int lane = tid & 31;

  // ---- stage transposed f16 weights into LDS (coalesced global reads) ----
  for (int i = tid; i < D_EDGE * IN_DIM; i += 256) {
    const int k = i >> 6;   // 0..191
    const int n = i & 63;
    sW1[n * IN_DIM + k] = (_Float16)W1[i];
  }
  for (int i = tid; i < D_EDGE * D_EDGE; i += 256) {
    const int k = i >> 6;   // 0..63
    const int n = i & 63;
    sW2[n * D_EDGE + k] = (_Float16)W2[i];
  }
  __syncthreads();

  const int m    = lane & 15;        // tile row this lane feeds (A fragments)
  const int half = lane >> 4;        // lane group 0 / 1
  const int c0   = half * 8;         // A-fragment K sub-chunk base
  const int kB   = half * 8;         // B-fragment K sub-chunk base
  const int ncol = lane & 15;        // output column within N-tile (B/C frags)

  const int eBase = blockIdx.x * EDGES_PER_BLOCK + wave * 16;
  int e = eBase + m;
  if (e >= E) e = E - 1;             // tail clamp (no EXEC divergence at WMMA)

  const int s = src_idxs[e];
  const int d = dst_idxs[e];
  const float* sp = node_scalars + (size_t)s * D_NODE;
  const float* dp = node_scalars + (size_t)d * D_NODE;
  const float* ep = edge_feats   + (size_t)e * D_EDGE;

  // ---- A fragments, GEMM1: 16x192 tile = 6 K-blocks of 32 ----
  v16h A[6];
  A[0] = pack2(sp + c0,      sp + c0 + 16);
  A[1] = pack2(sp + c0 + 32, sp + c0 + 48);
  A[2] = pack2(dp + c0,      dp + c0 + 16);
  A[3] = pack2(dp + c0 + 32, dp + c0 + 48);
  A[4] = pack2(ep + c0,      ep + c0 + 16);
  A[5] = pack2(ep + c0 + 32, ep + c0 + 48);

  _Float16* hTile = &sH1[wave * 16 * H1_LD];

  // ---- GEMM1 + bias + silu -> f16 h1 tile in LDS ----
#pragma unroll
  for (int nt = 0; nt < 4; ++nt) {
    const int n = nt * 16 + ncol;
    v8f acc = {};
    const _Float16* wrow = sW1 + n * IN_DIM + kB;
#pragma unroll
    for (int kb = 0; kb < 6; ++kb) {
      v16h B = __builtin_shufflevector(*(const v8h*)(wrow + kb * 32), *(const v8h*)(wrow + kb * 32 + 16), 0, 1, 2, 3, 4, 5, 6, 7, 8, 9, 10, 11, 12, 13, 14, 15);
      acc = __builtin_amdgcn_wmma_f32_16x16x32_f16(false, A[kb], false, B,
                                                   (short)0, acc, false, false);
    }
    const float bias = b1[n];
#pragma unroll
    for (int r = 0; r < 8; ++r) {
      const int M = r + half * 8;
      hTile[M * H1_LD + n] = (_Float16)silu_f(acc[r] + bias);
    }
  }
  __syncthreads();

  // ---- A fragments, GEMM2: 16x64 tile = 2 K-blocks of 32 (from LDS) ----
  v16h A2[2];
  const _Float16* hrow = hTile + m * H1_LD;
#pragma unroll
  for (int kb = 0; kb < 2; ++kb) {
    v8h lo = *(const v8h*)(hrow + kb * 32 + c0);
    v8h hi = *(const v8h*)(hrow + kb * 32 + 16 + c0);
#pragma unroll
    for (int j = 0; j < 8; ++j) { A2[kb][j] = lo[j]; A2[kb][8 + j] = hi[j]; }
  }

  // ---- GEMM2 ----
  v8f acc2[4];
#pragma unroll
  for (int nt = 0; nt < 4; ++nt) {
    const int n = nt * 16 + ncol;
    v8f acc = {};
    const _Float16* wrow = sW2 + n * D_EDGE + kB;
#pragma unroll
    for (int kb = 0; kb < 2; ++kb) {
      v16h B = __builtin_shufflevector(*(const v8h*)(wrow + kb * 32), *(const v8h*)(wrow + kb * 32 + 16), 0, 1, 2, 3, 4, 5, 6, 7, 8, 9, 10, 11, 12, 13, 14, 15);
      acc = __builtin_amdgcn_wmma_f32_16x16x32_f16(false, A2[kb], false, B,
                                                   (short)0, acc, false, false);
    }
    acc2[nt] = acc;
  }

  // per-column constants for epilogue
  float bb[4], gg[4], be[4];
#pragma unroll
  for (int nt = 0; nt < 4; ++nt) {
    const int n = nt * 16 + ncol;
    bb[nt] = b2[n];
    gg[nt] = ln_gamma[n];
    be[nt] = ln_beta[n];
  }

  // ---- residual + LayerNorm (row reductions via shfl_xor within 16-lane half) ----
#pragma unroll
  for (int r = 0; r < 8; ++r) {
    const int M = r + half * 8;
    int er = eBase + M;
    const bool valid = (er < E);
    if (!valid) er = E - 1;
    const size_t rowOff = (size_t)er * D_EDGE;
    float y[4];
    float sum = 0.0f, sq = 0.0f;
#pragma unroll
    for (int nt = 0; nt < 4; ++nt) {
      const int n = nt * 16 + ncol;
      const float v = edge_feats[rowOff + n] + silu_f(acc2[nt][r] + bb[nt]);
      y[nt] = v;
      sum += v;
      sq  += v * v;
    }
#pragma unroll
    for (int off = 1; off < 16; off <<= 1) {
      sum += __shfl_xor(sum, off, 32);
      sq  += __shfl_xor(sq,  off, 32);
    }
    const float mean = sum * (1.0f / 64.0f);
    const float var  = sq * (1.0f / 64.0f) - mean * mean;
    const float rstd = rsqrtf(var + 1e-5f);
    if (valid) {
#pragma unroll
      for (int nt = 0; nt < 4; ++nt) {
        const int n = nt * 16 + ncol;
        out[rowOff + n] = (y[nt] - mean) * rstd * gg[nt] + be[nt];
      }
    }
  }
}

extern "C" void kernel_launch(void* const* d_in, const int* in_sizes, int n_in,
                              void* d_out, int out_size, void* d_ws, size_t ws_size,
                              hipStream_t stream) {
  const float* node_scalars = (const float*)d_in[0];
  const float* edge_feats   = (const float*)d_in[1];
  const int*   src_idxs     = (const int*)d_in[2];
  const int*   dst_idxs     = (const int*)d_in[3];
  const float* W1           = (const float*)d_in[4];
  const float* b1           = (const float*)d_in[5];
  const float* W2           = (const float*)d_in[6];
  const float* b2           = (const float*)d_in[7];
  const float* ln_gamma     = (const float*)d_in[8];
  const float* ln_beta      = (const float*)d_in[9];
  float* out = (float*)d_out;

  const int E = in_sizes[2];  // number of edges (src_idxs length)
  const int blocks = (E + EDGES_PER_BLOCK - 1) / EDGES_PER_BLOCK;

  edge_mlp_ln_kernel<<<blocks, 256, 0, stream>>>(
      node_scalars, edge_feats, src_idxs, dst_idxs,
      W1, b1, W2, b2, ln_gamma, ln_beta, out, E);
}
